// BiGCN_graphcl_29111288332560
// MI455X (gfx1250) — matrix operand order repaired and checked by reading, not compiled, not run on hardware
//
#include <hip/hip_runtime.h>
#include <hip/hip_bf16.h>
#include <math.h>

typedef __bf16 bf16_t;
typedef __attribute__((ext_vector_type(16))) __bf16 v16bf;
typedef __attribute__((ext_vector_type(8)))  float  v8f;

#define GG   1000
#define SS   100
#define FIN  512
#define FOUT 64
#define HIDN 128
#define DIMN 128
#define LPITCH 48   // bf16 elems per LDS row (96B, 16B-aligned, padded vs 32)

// ---------------------------------------------------------------------------
// Generic WMMA GEMM:  C[M,Nn] = EPI( A[M,K] @ B[K,Nn] (+bias) )
//   A: fp32 (AF32=true, converted on the fly) or bf16 row-major
//   GATHER: row r of A is taken from rowidx[r]
//   Bt: bf16, TRANSPOSED (Nn x K row-major) so LDS fragment loads are contiguous
//   EPI: 0=none  1=+bias  2=+bias,leakyReLU(0.01)  3=+bias,ReLU
//        4=+bias,leakyReLU, then z = (1-a)*(y*pmul) + a*(y+padd)   (adapter fuse)
// Tile: 128(M) x 64(N), 8 waves (wave32), each wave 32x32 via 4x v_wmma 16x16x32.
// Double-buffered LDS: global loads for tile k+1 overlap WMMA on tile k,
// one barrier per K-step.
// ---------------------------------------------------------------------------
template<int EPI, bool AF32, bool GATHER, typename OutT>
__global__ __launch_bounds__(256)
void gemm_wmma(const void* __restrict__ Aptr, const bf16_t* __restrict__ Bt,
               const float* __restrict__ bias, OutT* __restrict__ C,
               int M, int Nn, int K,
               const int* __restrict__ rowidx,
               const int* __restrict__ batch,
               const float* __restrict__ alpha,
               const float* __restrict__ pmul,
               const float* __restrict__ padd)
{
  __shared__ __align__(16) bf16_t lsA[2 * 128 * LPITCH];
  __shared__ __align__(16) bf16_t lsB[2 * 64  * LPITCH];

  const int tid  = threadIdx.x;
  const int lane = tid & 31;
  const int wave = tid >> 5;
  const int wm   = wave >> 1;     // 0..3  -> M sub-tile
  const int wn   = wave & 1;      // 0..1  -> N sub-tile
  const int m0   = blockIdx.y * 128;
  const int n0   = blockIdx.x * 64;

  // ---- k-invariant staging descriptors (hoisted out of the K loop) ----
  bool aOK[4]; int aRow[4], aCC[4]; const char* aPtr[4];
  #pragma unroll
  for (int it = 0; it < 4; ++it) {
    int q   = tid + it * 256;          // 0..1023
    int row = q >> 3;
    int cc  = (q & 7) << 2;
    int r   = m0 + row;
    aOK[it] = (r < M);
    long ra = 0;
    if (aOK[it]) ra = GATHER ? (long)rowidx[r] : (long)r;
    aPtr[it] = (const char*)Aptr + (size_t)ra * K * (AF32 ? 4 : 2);
    aRow[it] = row; aCC[it] = cc;
  }
  bool bOK[2]; int bRow[2], bCC[2]; const bf16_t* bPtr[2];
  #pragma unroll
  for (int it = 0; it < 2; ++it) {
    int q   = tid + it * 256;          // 0..511
    int row = q >> 3;
    int cc  = (q & 7) << 2;
    int n   = n0 + row;
    bOK[it] = (n < Nn);
    bPtr[it] = Bt + (size_t)(bOK[it] ? n : 0) * K;
    bRow[it] = row; bCC[it] = cc;
  }

  v8f acc[2][2];
  const v8f z8 = {0.f,0.f,0.f,0.f,0.f,0.f,0.f,0.f};
  #pragma unroll
  for (int i = 0; i < 2; ++i)
    #pragma unroll
    for (int j = 0; j < 2; ++j) acc[i][j] = z8;

  union FragU { uint4 q[2]; v16bf v; };

  float4 aF[4]; uint2 aH[4]; uint2 bH[2];

  auto loadStage = [&](int k0) {
    #pragma unroll
    for (int it = 0; it < 4; ++it) {
      if (AF32) {
        const float4 zz = {0.f,0.f,0.f,0.f};
        aF[it] = aOK[it] ? *(const float4*)(aPtr[it] + (size_t)(k0 + aCC[it]) * 4) : zz;
      } else {
        const uint2 zz = {0u,0u};
        aH[it] = aOK[it] ? *(const uint2*)(aPtr[it] + (size_t)(k0 + aCC[it]) * 2) : zz;
      }
    }
    #pragma unroll
    for (int it = 0; it < 2; ++it) {
      const uint2 zz = {0u,0u};
      bH[it] = bOK[it] ? *(const uint2*)(bPtr[it] + k0 + bCC[it]) : zz;
    }
  };
  auto storeStage = [&](int buf) {
    #pragma unroll
    for (int it = 0; it < 4; ++it) {
      bf16_t* d = &lsA[buf * 128 * LPITCH + aRow[it] * LPITCH + aCC[it]];
      if (AF32) {
        d[0] = (bf16_t)aF[it].x; d[1] = (bf16_t)aF[it].y;
        d[2] = (bf16_t)aF[it].z; d[3] = (bf16_t)aF[it].w;
      } else {
        *(uint2*)d = aH[it];
      }
    }
    #pragma unroll
    for (int it = 0; it < 2; ++it)
      *(uint2*)&lsB[buf * 64 * LPITCH + bRow[it] * LPITCH + bCC[it]] = bH[it];
  };

  // ---- prologue: stage tile 0 ----
  loadStage(0);
  storeStage(0);
  __syncthreads();

  int buf = 0;
  for (int k0 = 0; k0 < K; k0 += 32) {
    const bool hasNext = (k0 + 32) < K;
    if (hasNext) loadStage(k0 + 32);      // global loads in flight during WMMA

    // ---- fragments from current buffer (ISA 7.12.2 layouts) ----
    v16bf a[2], b[2];
    #pragma unroll
    for (int i = 0; i < 2; ++i) {
      FragU u;
      int arow = wm * 32 + i * 16 + (lane & 15);
      int akb  = (lane >> 4) * 8;                 // K interleave for A
      const char* pa = (const char*)&lsA[buf * 128 * LPITCH + arow * LPITCH + akb];
      u.q[0] = *(const uint4*)pa;                 // K = akb .. akb+7
      u.q[1] = *(const uint4*)(pa + 32);          // K = akb+16 .. akb+23
      a[i] = u.v;
    }
    #pragma unroll
    for (int j = 0; j < 2; ++j) {
      FragU u;
      int brow = wn * 32 + j * 16 + (lane & 15);
      int bkb  = (lane >> 4) * 8;                 // K interleave for B
      const char* pb = (const char*)&lsB[buf * 64 * LPITCH + brow * LPITCH + bkb];
      u.q[0] = *(const uint4*)pb;                 // K = bkb .. bkb+7
      u.q[1] = *(const uint4*)(pb + 32);          // K = bkb+16 .. bkb+23
      b[j] = u.v;
    }
    #pragma unroll
    for (int i = 0; i < 2; ++i)
      #pragma unroll
      for (int j = 0; j < 2; ++j)
        acc[i][j] = __builtin_amdgcn_wmma_f32_16x16x32_bf16(
            false, a[i], false, b[j], (short)0, acc[i][j], false, false);

    if (hasNext) storeStage(buf ^ 1);     // write next tile to other buffer
    __syncthreads();
    buf ^= 1;
  }

  // ---- epilogue + store (C layout: lane n=l&15, row = 8*(l>>4)+v) ----
  #pragma unroll
  for (int i = 0; i < 2; ++i) {
    #pragma unroll
    for (int j = 0; j < 2; ++j) {
      int rbase = m0 + wm * 32 + i * 16 + ((lane >> 4) << 3);
      int c     = n0 + wn * 32 + j * 16 + (lane & 15);
      if (c < Nn) {
        float bv = (EPI != 0 && bias) ? bias[c] : 0.f;
        #pragma unroll
        for (int v = 0; v < 8; ++v) {
          int r = rbase + v;
          if (r < M) {
            float x = acc[i][j][v] + bv;
            if (EPI == 2 || EPI == 4) x = (x > 0.f) ? x : 0.01f * x;
            if (EPI == 3)             x = fmaxf(x, 0.f);
            if (EPI == 4) {
              int g = batch[r];
              float al = alpha[g];
              float pm = pmul[(size_t)g * Nn + c];
              float pa = padd[(size_t)g * Nn + c];
              x = (1.f - al) * (x * pm) + al * (x + pa);
            }
            C[(size_t)r * Nn + c] = (OutT)x;
          }
        }
      }
    }
  }
}

// ---------------------------------------------------------------------------
// Support kernels
// ---------------------------------------------------------------------------
__global__ void k_fill(float* p, float v, int n) {
  int i = blockIdx.x * blockDim.x + threadIdx.x;
  if (i < n) p[i] = v;
}
__global__ void k_count_sizes(const int* __restrict__ batch, float* sz, int n) {
  int i = blockIdx.x * blockDim.x + threadIdx.x;
  if (i < n) atomicAdd(&sz[batch[i]], 1.0f);
}
__global__ void k_scan_starts(const float* __restrict__ sz, int* starts, int g) {
  __shared__ int sh[1024];
  int t = threadIdx.x;
  int v = (t < g) ? (int)sz[t] : 0;
  sh[t] = v;
  __syncthreads();
  for (int off = 1; off < 1024; off <<= 1) {
    int x = (t >= off) ? sh[t - off] : 0;
    __syncthreads();
    sh[t] += x;
    __syncthreads();
  }
  if (t < g) starts[t] = sh[t] - v;   // exclusive scan
}
__global__ void k_edges(const int* __restrict__ src, const int* __restrict__ dst,
                        const int* __restrict__ batch, const int* __restrict__ starts,
                        float* degTD, float* degBU, float* cnt, int E) {
  int e = blockIdx.x * blockDim.x + threadIdx.x;
  if (e >= E) return;
  int s = src[e], d = dst[e];
  atomicAdd(&degTD[d], 1.0f);
  atomicAdd(&degBU[s], 1.0f);
  int bs = batch[s];
  if (bs == batch[d] && s == starts[bs]) atomicAdd(&cnt[bs], 1.0f);
}
__global__ void k_rsqrt(float* p, int n) {
  int i = blockIdx.x * blockDim.x + threadIdx.x;
  if (i < n) p[i] = rsqrtf(p[i]);
}
__global__ void k_alpha(const float* cnt, const float* sz, float* al, int g) {
  int i = blockIdx.x * blockDim.x + threadIdx.x;
  if (i < g) {
    float t = (cnt[i] / sz[i] - 0.3f) / 0.1f;
    al[i] = 1.f / (1.f + expf(-t));
  }
}
__global__ void k_cvt_wT(const float* __restrict__ W, bf16_t* Wt, int K, int Nn) {
  int idx = blockIdx.x * blockDim.x + threadIdx.x;
  if (idx >= K * Nn) return;
  int k = idx / Nn, n = idx - k * Nn;
  Wt[(size_t)n * K + k] = (bf16_t)W[idx];
}
__global__ void k_reduce2(const float* __restrict__ h, int n, float* acc) {
  __shared__ float s1[256], s2[256];
  float a = 0.f, b = 0.f;
  for (int i = blockIdx.x * blockDim.x + threadIdx.x; i < n; i += gridDim.x * blockDim.x) {
    float v = h[i]; a += v; b += v * v;
  }
  s1[threadIdx.x] = a; s2[threadIdx.x] = b;
  __syncthreads();
  for (int off = 128; off > 0; off >>= 1) {
    if ((int)threadIdx.x < off) { s1[threadIdx.x] += s1[threadIdx.x + off]; s2[threadIdx.x] += s2[threadIdx.x + off]; }
    __syncthreads();
  }
  if (threadIdx.x == 0) { atomicAdd(&acc[0], s1[0]); atomicAdd(&acc[1], s2[0]); }
}
__global__ void k_ln_stats(float* acc, float n) {
  if (blockIdx.x == 0 && threadIdx.x == 0) {
    float mean = acc[0] / n;
    float ms   = acc[1] / n - mean * mean;
    acc[2] = mean;
    acc[3] = 1.f / (sqrtf(fmaxf(ms, 0.f)) + 1e-5f);
  }
}
__global__ void k_ln_tanh(const float* __restrict__ h, const float* __restrict__ st,
                          const float* __restrict__ lw, const float* __restrict__ lb,
                          bf16_t* out, int n, int D) {
  int i = blockIdx.x * blockDim.x + threadIdx.x;
  if (i < n) {
    int c = i % D;
    float v = (h[i] - st[2]) * st[3];
    out[i] = (bf16_t)tanhf(v * lw[c] + lb[c]);
  }
}
__global__ void k_gcn_init(const float* __restrict__ h, const float* __restrict__ dinv,
                           const float* __restrict__ bias, float* out, int n, int C) {
  size_t idx = (size_t)blockIdx.x * blockDim.x + threadIdx.x;
  if (idx >= (size_t)n * C) return;
  int i = (int)(idx / C); int c = (int)(idx - (size_t)i * C);
  float dv = dinv[i];
  out[idx] = h[idx] * dv * dv + bias[c];
}
__global__ void k_gcn_edge(const int* __restrict__ es, const int* __restrict__ ed,
                           const float* __restrict__ h, const float* __restrict__ dinv,
                           float* out, int E, int Cq, int C) {
  int idx = blockIdx.x * blockDim.x + threadIdx.x;
  if (idx >= E * Cq) return;
  int e = idx / Cq, q = idx - e * Cq;
  int s = es[e], d = ed[e];
  float w = dinv[s] * dinv[d];
  const float4 hv = *(const float4*)(h + (size_t)s * C + q * 4);
  float* o = out + (size_t)d * C + q * 4;
  atomicAdd(o + 0, hv.x * w);
  atomicAdd(o + 1, hv.y * w);
  atomicAdd(o + 2, hv.z * w);
  atomicAdd(o + 3, hv.w * w);
}
__global__ void k_relu_cvt(const float* __restrict__ in, bf16_t* out, int n) {
  int i = blockIdx.x * blockDim.x + threadIdx.x;
  if (i < n) out[i] = (bf16_t)fmaxf(in[i], 0.f);
}
__global__ void k_cvt_b(const float* __restrict__ in, bf16_t* out, int n) {
  int i = blockIdx.x * blockDim.x + threadIdx.x;
  if (i < n) out[i] = (bf16_t)in[i];
}
__global__ void k_segsum(const float* __restrict__ h, const int* __restrict__ starts,
                         const float* __restrict__ sz, float* out,
                         int C, int stride, int offc) {
  int g = blockIdx.x, c = threadIdx.x;
  int st = starts[g]; int n = (int)sz[g];
  float s = 0.f;
  for (int r = 0; r < n; ++r) s += h[(size_t)(st + r) * C + c];
  out[(size_t)g * stride + offc + c] = s;
}

// ---------------------------------------------------------------------------
extern "C" void kernel_launch(void* const* d_in, const int* in_sizes, int n_in,
                              void* d_out, int out_size, void* d_ws, size_t ws_size,
                              hipStream_t stream) {
  const float* x        = (const float*)d_in[0];
  const int*   ei       = (const int*)d_in[1];
  const int*   batch    = (const int*)d_in[2];
  const float* adapterW = (const float*)d_in[3];
  const float* adapterB = (const float*)d_in[4];
  const float* p1w1 = (const float*)d_in[5],  *p1b1 = (const float*)d_in[6];
  const float* p1lw = (const float*)d_in[7],  *p1lb = (const float*)d_in[8];
  const float* p1w2 = (const float*)d_in[9],  *p1b2 = (const float*)d_in[10];
  const float* p2w1 = (const float*)d_in[11], *p2b1 = (const float*)d_in[12];
  const float* p2lw = (const float*)d_in[13], *p2lb = (const float*)d_in[14];
  const float* p2w2 = (const float*)d_in[15], *p2b2 = (const float*)d_in[16];
  const float* td1w = (const float*)d_in[17], *td1b = (const float*)d_in[18];
  const float* td2w = (const float*)d_in[19], *td2b = (const float*)d_in[20];
  const float* bu1w = (const float*)d_in[21], *bu1b = (const float*)d_in[22];
  const float* bu2w = (const float*)d_in[23], *bu2b = (const float*)d_in[24];
  const float* pjw1 = (const float*)d_in[25], *pjb1 = (const float*)d_in[26];
  const float* pjw2 = (const float*)d_in[27], *pjb2 = (const float*)d_in[28];

  const int NN = in_sizes[0] / FIN;        // 100000
  const int E  = in_sizes[1] / 2;          // 99000
  const int*   src = ei;
  const int*   dst = ei + E;

  // ---- workspace layout ----
  char* ws = (char*)d_ws;
  size_t off = 0;
  auto alloc = [&](size_t bytes) -> char* {
    char* p = ws + off;
    off = (off + bytes + 255) & ~(size_t)255;
    return p;
  };
  bf16_t* wtA   = (bf16_t*)alloc((size_t)FIN * FIN * 2);
  bf16_t* wtTD1 = (bf16_t*)alloc((size_t)FIN * HIDN * 2);
  bf16_t* wtTD2 = (bf16_t*)alloc((size_t)HIDN * FOUT * 2);
  bf16_t* wtBU1 = (bf16_t*)alloc((size_t)FIN * HIDN * 2);
  bf16_t* wtBU2 = (bf16_t*)alloc((size_t)HIDN * FOUT * 2);
  bf16_t* wtP1A = (bf16_t*)alloc((size_t)FIN * DIMN * 2);
  bf16_t* wtP1B = (bf16_t*)alloc((size_t)DIMN * FIN * 2);
  bf16_t* wtP2A = (bf16_t*)alloc((size_t)FIN * DIMN * 2);
  bf16_t* wtP2B = (bf16_t*)alloc((size_t)DIMN * FIN * 2);
  bf16_t* wtPJ1 = (bf16_t*)alloc((size_t)128 * 256 * 2);
  bf16_t* wtPJ2 = (bf16_t*)alloc((size_t)256 * 128 * 2);
  float* sizesF = (float*)alloc((size_t)GG * 4);
  int*   starts = (int*)  alloc((size_t)GG * 4);
  float* cntF   = (float*)alloc((size_t)GG * 4);
  float* alphaF = (float*)alloc((size_t)GG * 4);
  float* dinvTD = (float*)alloc((size_t)NN * 4);
  float* dinvBU = (float*)alloc((size_t)NN * 4);
  float* yroot  = (float*)alloc((size_t)GG * FIN * 4);
  float* pmid   = (float*)alloc((size_t)GG * DIMN * 4);
  bf16_t* midbf = (bf16_t*)alloc((size_t)GG * DIMN * 2);
  float* lnacc  = (float*)alloc(256);
  float* pmul   = (float*)alloc((size_t)GG * FIN * 4);
  float* padd   = (float*)alloc((size_t)GG * FIN * 4);
  bf16_t* zbf   = (bf16_t*)alloc((size_t)NN * FIN * 2);
  float* h1     = (float*)alloc((size_t)NN * HIDN * 4);
  float* g1     = (float*)alloc((size_t)NN * HIDN * 4);
  bf16_t* h1bf  = (bf16_t*)alloc((size_t)NN * HIDN * 2);
  float* hcat   = (float*)alloc((size_t)GG * 128 * 4);
  bf16_t* hcatb = (bf16_t*)alloc((size_t)GG * 128 * 2);
  bf16_t* hp    = (bf16_t*)alloc((size_t)GG * 256 * 2);
  float* h2 = h1;  // overlays: h1/g1 free once h1bf built
  float* g2 = g1;

  const dim3 blk(256);
  auto cdiv = [](long long a, long long b) { return (unsigned)((a + b - 1) / b); };
  auto ggrid = [&](int M, int Nn) { return dim3(cdiv(Nn, 64), cdiv(M, 128)); };

  // ---- graph statistics ----
  k_fill<<<cdiv(GG,256), blk, 0, stream>>>(sizesF, 0.f, GG);
  k_fill<<<cdiv(GG,256), blk, 0, stream>>>(cntF, 0.f, GG);
  k_fill<<<cdiv(NN,256), blk, 0, stream>>>(dinvTD, 1.f, NN);
  k_fill<<<cdiv(NN,256), blk, 0, stream>>>(dinvBU, 1.f, NN);
  k_count_sizes<<<cdiv(NN,256), blk, 0, stream>>>(batch, sizesF, NN);
  k_scan_starts<<<1, 1024, 0, stream>>>(sizesF, starts, GG);
  k_edges<<<cdiv(E,256), blk, 0, stream>>>(src, dst, batch, starts, dinvTD, dinvBU, cntF, E);
  k_rsqrt<<<cdiv(NN,256), blk, 0, stream>>>(dinvTD, NN);
  k_rsqrt<<<cdiv(NN,256), blk, 0, stream>>>(dinvBU, NN);
  k_alpha<<<cdiv(GG,256), blk, 0, stream>>>(cntF, sizesF, alphaF, GG);

  // ---- weight transposed-bf16 conversions ----
  k_cvt_wT<<<cdiv((size_t)FIN*FIN,256), blk, 0, stream>>>(adapterW, wtA, FIN, FIN);
  k_cvt_wT<<<cdiv((size_t)FIN*HIDN,256), blk, 0, stream>>>(td1w, wtTD1, FIN, HIDN);
  k_cvt_wT<<<cdiv((size_t)HIDN*FOUT,256), blk, 0, stream>>>(td2w, wtTD2, HIDN, FOUT);
  k_cvt_wT<<<cdiv((size_t)FIN*HIDN,256), blk, 0, stream>>>(bu1w, wtBU1, FIN, HIDN);
  k_cvt_wT<<<cdiv((size_t)HIDN*FOUT,256), blk, 0, stream>>>(bu2w, wtBU2, HIDN, FOUT);
  k_cvt_wT<<<cdiv((size_t)FIN*DIMN,256), blk, 0, stream>>>(p1w1, wtP1A, FIN, DIMN);
  k_cvt_wT<<<cdiv((size_t)DIMN*FIN,256), blk, 0, stream>>>(p1w2, wtP1B, DIMN, FIN);
  k_cvt_wT<<<cdiv((size_t)FIN*DIMN,256), blk, 0, stream>>>(p2w1, wtP2A, FIN, DIMN);
  k_cvt_wT<<<cdiv((size_t)DIMN*FIN,256), blk, 0, stream>>>(p2w2, wtP2B, DIMN, FIN);
  k_cvt_wT<<<cdiv(128*256,256), blk, 0, stream>>>(pjw1, wtPJ1, 128, 256);
  k_cvt_wT<<<cdiv(256*128,256), blk, 0, stream>>>(pjw2, wtPJ2, 256, 128);

  // ---- root adapter (gathered rows) : yroot = lrelu(x[starts] @ Wa + b) ----
  gemm_wmma<2, true, true, float><<<ggrid(GG, FIN), blk, 0, stream>>>(
      x, wtA, adapterB, yroot, GG, FIN, FIN, starts, nullptr, nullptr, nullptr, nullptr);

  // ---- prompt heads: tanh(graphLN(yroot@w1+b1)) @ w2 + b2 ----
  for (int head = 0; head < 2; ++head) {
    const bf16_t* wA = head ? wtP2A : wtP1A;
    const bf16_t* wB = head ? wtP2B : wtP1B;
    const float *b1 = head ? p2b1 : p1b1, *b2 = head ? p2b2 : p1b2;
    const float *lw = head ? p2lw : p1lw, *lb = head ? p2lb : p1lb;
    float* outP = head ? padd : pmul;
    gemm_wmma<1, true, false, float><<<ggrid(GG, DIMN), blk, 0, stream>>>(
        yroot, wA, b1, pmid, GG, DIMN, FIN, nullptr, nullptr, nullptr, nullptr, nullptr);
    k_fill<<<1, 32, 0, stream>>>(lnacc, 0.f, 4);
    k_reduce2<<<cdiv(GG*DIMN,256), blk, 0, stream>>>(pmid, GG*DIMN, lnacc);
    k_ln_stats<<<1, 32, 0, stream>>>(lnacc, (float)(GG*DIMN));
    k_ln_tanh<<<cdiv(GG*DIMN,256), blk, 0, stream>>>(pmid, lnacc, lw, lb, midbf, GG*DIMN, DIMN);
    gemm_wmma<1, false, false, float><<<ggrid(GG, FIN), blk, 0, stream>>>(
        midbf, wB, b2, outP, GG, FIN, DIMN, nullptr, nullptr, nullptr, nullptr, nullptr);
  }

  // ---- fused adapter + prompt modulation -> z (bf16) ----
  gemm_wmma<4, true, false, bf16_t><<<ggrid(NN, FIN), blk, 0, stream>>>(
      x, wtA, adapterB, zbf, NN, FIN, FIN, nullptr, batch, alphaF, pmul, padd);

  // ---- TD / BU GCN stacks ----
  for (int dir = 0; dir < 2; ++dir) {
    const bf16_t* w1 = dir ? wtBU1 : wtTD1;
    const bf16_t* w2 = dir ? wtBU2 : wtTD2;
    const float *bb1 = dir ? bu1b : td1b, *bb2 = dir ? bu2b : td2b;
    const float* dinv = dir ? dinvBU : dinvTD;
    const int* es = dir ? dst : src;   // BU aggregates flipped edges
    const int* ed = dir ? src : dst;
    // layer 1: h = z @ w1 ; out = agg + h*dinv^2 + b ; relu ; -> bf16
    gemm_wmma<0, false, false, float><<<ggrid(NN, HIDN), blk, 0, stream>>>(
        zbf, w1, nullptr, h1, NN, HIDN, FIN, nullptr, nullptr, nullptr, nullptr, nullptr);
    k_gcn_init<<<cdiv((size_t)NN*HIDN,256), blk, 0, stream>>>(h1, dinv, bb1, g1, NN, HIDN);
    k_gcn_edge<<<cdiv((size_t)E*(HIDN/4),256), blk, 0, stream>>>(es, ed, h1, dinv, g1, E, HIDN/4, HIDN);
    k_relu_cvt<<<cdiv((size_t)NN*HIDN,256), blk, 0, stream>>>(g1, h1bf, NN*HIDN);
    // layer 2
    gemm_wmma<0, false, false, float><<<ggrid(NN, FOUT), blk, 0, stream>>>(
        h1bf, w2, nullptr, h2, NN, FOUT, HIDN, nullptr, nullptr, nullptr, nullptr, nullptr);
    k_gcn_init<<<cdiv((size_t)NN*FOUT,256), blk, 0, stream>>>(h2, dinv, bb2, g2, NN, FOUT);
    k_gcn_edge<<<cdiv((size_t)E*(FOUT/4),256), blk, 0, stream>>>(es, ed, h2, dinv, g2, E, FOUT/4, FOUT);
    // readout into [BU | TD] concat buffer
    k_segsum<<<GG, FOUT, 0, stream>>>(g2, starts, sizesF, hcat, FOUT, 128, dir ? 0 : 64);
  }

  // ---- projection head ----
  k_cvt_b<<<cdiv(GG*128,256), blk, 0, stream>>>(hcat, hcatb, GG*128);
  gemm_wmma<3, false, false, bf16_t><<<ggrid(GG, 256), blk, 0, stream>>>(
      hcatb, wtPJ1, pjb1, hp, GG, 256, 128, nullptr, nullptr, nullptr, nullptr, nullptr);
  gemm_wmma<1, false, false, float><<<ggrid(GG, 128), blk, 0, stream>>>(
      hp, wtPJ2, pjb2, (float*)d_out, GG, 128, 256, nullptr, nullptr, nullptr, nullptr, nullptr);

  (void)n_in; (void)out_size; (void)ws_size;
}
